// MaskedAttention_57552561766491
// MI455X (gfx1250) — matrix operand order repaired and checked by reading, not compiled, not run on hardware
//
#include <hip/hip_runtime.h>

// MaskedAttention: out = softmax_masked(QK^T/sqrt(D) - |i-j|) @ V, returns (out, attention)
// B=16, S=1024, D=512, fp32 in/out; bf16 WMMA internally (memory-bound design point).

#define BB 16
#define SS 1024
#define DD 512

typedef __bf16 v16bf __attribute__((ext_vector_type(16)));
typedef __bf16 v8bf  __attribute__((ext_vector_type(8)));
typedef float  v8f   __attribute__((ext_vector_type(8)));
typedef float  v4f   __attribute__((ext_vector_type(4)));

__global__ __launch_bounds__(256) void attn_fused_kernel(
    const float* __restrict__ q, const float* __restrict__ k,
    const float* __restrict__ v, const int* __restrict__ rep,
    float* __restrict__ outp, float* __restrict__ attn)
{
    // probs slab: 16 rows x (1024+8) bf16, row stride 2064B = 129*16B (odd multiple of 16B)
    __shared__ __bf16 probs[16 * 1032];
    __shared__ float  redp[16][8];
    __shared__ float  redf[16];

    const int qtile = blockIdx.x;
    const int b     = blockIdx.y;
    const int tid   = threadIdx.x;
    const int wave  = tid >> 5;
    const int lane  = tid & 31;
    const int hi    = lane >> 4;   // half-wave select
    const int ln    = lane & 15;
    const int qr0   = qtile * 16;

    const float* qb = q + (size_t)b * SS * DD;
    const float* kb = k + (size_t)b * SS * DD;
    const float* vb = v + (size_t)b * SS * DD;
    const int*   rb = rep + (size_t)b * SS;
    float* outb  = outp + (size_t)b * SS * DD;
    float* attnb = attn + (size_t)b * SS * SS;

    const v8f vzero = {0.f,0.f,0.f,0.f,0.f,0.f,0.f,0.f};
    const float SCALE = 0.04419417382415922f; // 1/sqrt(512), folded into Q

    // ---------------- Phase 1: S = (Q/sqrt(D)) K^T, per-wave 8 column tiles ----------------
    v8f acc[8];
    #pragma unroll
    for (int i = 0; i < 8; ++i) acc[i] = vzero;

    #pragma unroll 1
    for (int dc = 0; dc < 16; ++dc) {
        const int d0 = dc * 32;
        // A fragment: lane ln -> row m=ln; K pattern {8*hi+0..7, 16+8*hi+0..7}
        v16bf afrag;
        {
            const float* ap = qb + (size_t)(qr0 + ln) * DD + d0 + 8 * hi;
            v4f a0 = *(const v4f*)(ap);
            v4f a1 = *(const v4f*)(ap + 4);
            v4f a2 = *(const v4f*)(ap + 16);
            v4f a3 = *(const v4f*)(ap + 20);
            #pragma unroll
            for (int j = 0; j < 4; ++j) {
                afrag[j]      = (__bf16)(a0[j] * SCALE);
                afrag[4 + j]  = (__bf16)(a1[j] * SCALE);
                afrag[8 + j]  = (__bf16)(a2[j] * SCALE);
                afrag[12 + j] = (__bf16)(a3[j] * SCALE);
            }
        }
        #pragma unroll
        for (int ct = 0; ct < 8; ++ct) {
            const int c0 = (wave * 8 + ct) * 16;
            // B fragment: lane -> column n=ln, K pattern {d0+8*hi+0..7, d0+16+8*hi+0..7}, i.e. kb[c0+ln][k]
            const float* kp = kb + (size_t)(c0 + ln) * DD + d0 + 8 * hi;
            v16bf bfrag;
            #pragma unroll
            for (int n = 0; n < 16; ++n) bfrag[n] = (__bf16)kp[n + (n & 8)];
            acc[ct] = __builtin_amdgcn_wmma_f32_16x16x32_bf16(
                false, afrag, false, bfrag, (short)0, acc[ct], false, false);
        }
    }

    // ---------------- Softmax (faithful to reference masked_softmax) ----------------
    int rmq[8];
    #pragma unroll
    for (int r = 0; r < 8; ++r) rmq[r] = rb[qr0 + r + 8 * hi];

    float mx[8];
    #pragma unroll
    for (int r = 0; r < 8; ++r) mx[r] = 0.0f;  // masked entries contribute 0 to the max

    #pragma unroll
    for (int ct = 0; ct < 8; ++ct) {
        const int colc = (wave * 8 + ct) * 16 + ln;
        const int rmk  = rb[colc];
        #pragma unroll
        for (int r = 0; r < 8; ++r) {
            const int qr = qr0 + r + 8 * hi;
            float s = acc[ct][r] - fabsf((float)(qr - colc));   // ALPHA = 1
            const bool mk = (colc < qr) && (rmq[r] != 0) && (rmk != 0);
            s = mk ? s : -1e30f;   // expf(-huge) == 0 => masked prob = 0
            acc[ct][r] = s;
            mx[r] = fmaxf(mx[r], s);
        }
    }
    #pragma unroll
    for (int r = 0; r < 8; ++r) {
        #pragma unroll
        for (int off = 1; off < 16; off <<= 1)
            mx[r] = fmaxf(mx[r], __shfl_xor(mx[r], off, 32));
    }
    if (ln == 0) {
        #pragma unroll
        for (int r = 0; r < 8; ++r) redp[r + 8 * hi][wave] = mx[r];
    }
    __syncthreads();
    if (tid < 16) {
        float m = redp[tid][0];
        #pragma unroll
        for (int w = 1; w < 8; ++w) m = fmaxf(m, redp[tid][w]);
        redf[tid] = m;
    }
    __syncthreads();
    float rowM[8];
    #pragma unroll
    for (int r = 0; r < 8; ++r) rowM[r] = redf[r + 8 * hi];

    float sm[8];
    #pragma unroll
    for (int r = 0; r < 8; ++r) sm[r] = 0.0f;
    #pragma unroll
    for (int ct = 0; ct < 8; ++ct) {
        #pragma unroll
        for (int r = 0; r < 8; ++r) {
            float p = __expf(acc[ct][r] - rowM[r]);
            acc[ct][r] = p;
            sm[r] += p;
        }
    }
    #pragma unroll
    for (int r = 0; r < 8; ++r) {
        #pragma unroll
        for (int off = 1; off < 16; off <<= 1)
            sm[r] += __shfl_xor(sm[r], off, 32);
    }
    if (ln == 0) {
        #pragma unroll
        for (int r = 0; r < 8; ++r) redp[r + 8 * hi][wave] = sm[r];
    }
    __syncthreads();
    if (tid < 16) {
        float s = 0.0f;
        #pragma unroll
        for (int w = 0; w < 8; ++w) s += redp[tid][w];
        s = s + ((s == 0.0f) ? 1.0f : 0.0f);   // reference: sums + (sums==0)
        redf[tid] = 1.0f / (s + 1e-20f);       // reference: / (sums + 1e-20)
    }
    __syncthreads();
    float inv[8];
    #pragma unroll
    for (int r = 0; r < 8; ++r) inv[r] = redf[r + 8 * hi];

    // Write attention (fp32, second output region) + bf16 probs slab in LDS
    #pragma unroll
    for (int ct = 0; ct < 8; ++ct) {
        const int colc = (wave * 8 + ct) * 16 + ln;
        #pragma unroll
        for (int r = 0; r < 8; ++r) {
            const int row = r + 8 * hi;
            const float a = acc[ct][r] * inv[r];
            attnb[(size_t)(qr0 + row) * SS + colc] = a;
            probs[row * 1032 + colc] = (__bf16)a;
        }
    }
    __syncthreads();

    // ---------------- Phase 2: out = P @ V, per-wave 4 n-tiles (64 cols of D) ----------------
    v8f acc2[4];
    #pragma unroll
    for (int i = 0; i < 4; ++i) acc2[i] = vzero;

    #pragma unroll 1
    for (int kc = 0; kc < 32; ++kc) {
        const int k0 = kc * 32;
        // A fragment from LDS probs: row m=ln, K pattern {k0+8*hi.., k0+16+8*hi..}
        const __bf16* pp = &probs[ln * 1032 + k0 + 8 * hi];
        v8bf lo = *(const v8bf*)(pp);
        v8bf hh = *(const v8bf*)(pp + 16);
        v16bf afrag = __builtin_shufflevector(lo, hh, 0,1,2,3,4,5,6,7,8,9,10,11,12,13,14,15);
        #pragma unroll
        for (int nt = 0; nt < 4; ++nt) {
            const int n0 = (wave * 4 + nt) * 16;
            // B fragment: lane -> V col n0+ln, K pattern {k0+8*hi+0..7, k0+16+8*hi+0..7}
            const float* vp = vb + (size_t)(k0 + 8 * hi) * DD + n0 + ln;
            v16bf bfrag;
            #pragma unroll
            for (int j = 0; j < 16; ++j) bfrag[j] = (__bf16)vp[(size_t)(j + (j & 8)) * DD];
            acc2[nt] = __builtin_amdgcn_wmma_f32_16x16x32_bf16(
                false, afrag, false, bfrag, (short)0, acc2[nt], false, false);
        }
    }

    #pragma unroll
    for (int nt = 0; nt < 4; ++nt) {
        const int n0 = (wave * 4 + nt) * 16;
        #pragma unroll
        for (int r = 0; r < 8; ++r) {
            outb[(size_t)(qr0 + r + 8 * hi) * DD + n0 + ln] = acc2[nt][r];
        }
    }
}

extern "C" void kernel_launch(void* const* d_in, const int* in_sizes, int n_in,
                              void* d_out, int out_size, void* d_ws, size_t ws_size,
                              hipStream_t stream) {
    const float* q   = (const float*)d_in[0];
    const float* k   = (const float*)d_in[1];
    const float* v   = (const float*)d_in[2];
    const int*   rep = (const int*)d_in[3];
    float* outp = (float*)d_out;                       // [B,S,D]
    float* attn = outp + (size_t)BB * SS * DD;         // [B,S,S], concatenated second output

    dim3 grid(SS / 16, BB);   // (q-tile, batch)
    attn_fused_kernel<<<grid, 256, 0, stream>>>(q, k, v, rep, outp, attn);
}
